// MultiHeadAttention_11699490914897
// MI455X (gfx1250) — matrix operand order repaired and checked by reading, not compiled, not run on hardware
//
#include <hip/hip_runtime.h>
#include <hip/hip_bf16.h>

typedef unsigned short u16;
typedef __attribute__((ext_vector_type(16))) __bf16 v16bf;
typedef __attribute__((ext_vector_type(8)))  float  v8f;
typedef __attribute__((ext_vector_type(4)))  int    v4i;

struct U16x16 { u16 v[16]; };

#if defined(__has_builtin)
#if __has_builtin(__builtin_amdgcn_global_load_async_to_lds_b128) && \
    __has_builtin(__builtin_amdgcn_s_wait_asynccnt)
#define USE_ASYNC_LDS 1
#endif
#if __has_builtin(__builtin_amdgcn_global_load_async_to_lds_b32) && \
    __has_builtin(__builtin_amdgcn_s_wait_asynccnt)
#define USE_ASYNC_LDS32 1
#endif
#endif
#ifndef USE_ASYNC_LDS
#define USE_ASYNC_LDS 0
#endif
#ifndef USE_ASYNC_LDS32
#define USE_ASYNC_LDS32 0
#endif

// address-space-qualified pointee typedefs for the async-to-LDS builtins
typedef __attribute__((address_space(1))) v4i g_v4i;
typedef __attribute__((address_space(3))) v4i l_v4i;
typedef __attribute__((address_space(1))) int g_i32;
typedef __attribute__((address_space(3))) int l_i32;

__device__ __forceinline__ u16 f2bf(float f) {
    unsigned int u = __builtin_bit_cast(unsigned int, f);
    unsigned int r = u + 0x7FFFu + ((u >> 16) & 1u);   // round-to-nearest-even
    return (u16)(r >> 16);
}

__device__ __forceinline__ v16bf pack16(const U16x16& s) {
    return __builtin_bit_cast(v16bf, s);
}

// ---- WMMA fragment loaders (wave32, v_wmma_f32_16x16x32_bf16) --------------
// A (16x32 bf16): lane = row (lane&15); 16 contiguous K [k0+h*16,+16)
__device__ __forceinline__ v16bf frag_a_global(const u16* A, long ld, int m0, int k0, int lane) {
    int h = lane >> 4;
    const u16* p = A + (long)(m0 + (lane & 15)) * ld + k0 + h * 16;
    U16x16 s;
#pragma unroll
    for (int i = 0; i < 16; ++i) s.v[i] = p[i];
    return pack16(s);
}

// B (32x16) for X @ W^T style access: lane = col (lane&15) == row of W; 16 contiguous K
__device__ __forceinline__ v16bf frag_b_global_bf16(const u16* Wt, long ld, int n0, int k0, int lane) {
    const u16* p = Wt + (long)(n0 + (lane & 15)) * ld + k0 + (lane >> 4) * 16;
    U16x16 s;
#pragma unroll
    for (int i = 0; i < 16; ++i) s.v[i] = p[i];
    return pack16(s);
}

__device__ __forceinline__ v16bf frag_b_global_f32(const float* W, long ld, int n0, int k0, int lane) {
    const float* p = W + (long)(n0 + (lane & 15)) * ld + k0 + (lane >> 4) * 16;
    U16x16 s;
#pragma unroll
    for (int i = 0; i < 16; ++i) s.v[i] = f2bf(p[i]);
    return pack16(s);
}

__device__ __forceinline__ v16bf frag_a_lds(const u16* As, int stride, int mrel, int k0, int lane) {
    int h = lane >> 4;
    const u16* p = As + (mrel + (lane & 15)) * stride + k0 + h * 16;
    U16x16 s;
#pragma unroll
    for (int i = 0; i < 16; ++i) s.v[i] = p[i];
    return pack16(s);
}

__device__ __forceinline__ v8f wmma_bf16(v16bf a, v16bf b, v8f c) {
    return __builtin_amdgcn_wmma_f32_16x16x32_bf16(false, a, false, b, (short)0, c, false, false);
}

// ---- GEMM: Out[M,N] = A[M,K] @ W[N,K]^T + bias (+ residual) ----------------
// block = 256 thr (8 waves), tile 16(M) x 128(N), waves split N.
template <bool A_BF16, bool OUT_BF16, bool RES>
__global__ __launch_bounds__(256) void gemm_xwT(const void* __restrict__ Ain,
                                                const float* __restrict__ W,
                                                const float* __restrict__ bias,
                                                const float* __restrict__ residual,
                                                void* __restrict__ Out,
                                                int M, int N, int K) {
    __shared__ u16 As[16][40];
    const int tid = threadIdx.x, lane = tid & 31, wave = tid >> 5;
    const int m0 = blockIdx.y * 16;
    const int n0 = blockIdx.x * 128 + wave * 16;
    v8f acc = {0.f, 0.f, 0.f, 0.f, 0.f, 0.f, 0.f, 0.f};

    for (int k0 = 0; k0 < K; k0 += 32) {
        {   // stage A tile (16x32) into LDS as bf16
            int r = tid >> 4;          // 0..15
            int c = (tid & 15) * 2;    // 0..30
            if constexpr (A_BF16) {
                const u16* p = (const u16*)Ain + (long)(m0 + r) * K + k0 + c;
#if USE_ASYNC_LDS32
                __builtin_amdgcn_global_load_async_to_lds_b32(
                    (g_i32*)p, (l_i32*)&As[r][c], 0, 0);
                __builtin_amdgcn_s_wait_asynccnt(0);
#else
                *(unsigned int*)&As[r][c] = *(const unsigned int*)p;
#endif
            } else {
                const float* p = (const float*)Ain + (long)(m0 + r) * K + k0 + c;
                As[r][c] = f2bf(p[0]); As[r][c + 1] = f2bf(p[1]);
                if (k0 + 32 < K) __builtin_prefetch(p + 32, 0, 0);  // global_prefetch_b8
            }
        }
        __syncthreads();
        v16bf a = frag_a_lds(&As[0][0], 40, 0, 0, lane);
        v16bf b = frag_b_global_f32(W, K, n0, k0, lane);
        acc = wmma_bf16(a, b, acc);
        __syncthreads();
    }

    const int n = n0 + (lane & 15);
    const int h = lane >> 4;
    const float bv = bias[n];
#pragma unroll
    for (int r = 0; r < 8; ++r) {
        int m = m0 + h * 8 + r;
        float v = acc[r] + bv;
        if constexpr (RES) v += residual[(long)m * N + n];
        if constexpr (OUT_BF16) ((u16*)Out)[(long)m * N + n] = f2bf(v);
        else                    ((float*)Out)[(long)m * N + n] = v;
    }
}

// ---- per-head transpose: Vt[b,h,d,j] = Vp[b,j,h*dk+d] ----------------------
__global__ __launch_bounds__(256) void transpose_v_kernel(const u16* __restrict__ Vp,
                                                          u16* __restrict__ Vt,
                                                          int S, int D, int H, int dk) {
    __shared__ u16 T[64][72];
    const int b = blockIdx.z, h = blockIdx.y, jt = blockIdx.x;
    const int t = threadIdx.x;
    const u16* src = Vp + ((long)b * S + (long)jt * 64) * D + (long)h * dk;
    {   // load 64(j) x 64(d) tile
        int j = t >> 2, c = (t & 3) * 16;
        const u16* p = src + (long)j * D + c;
#if USE_ASYNC_LDS
        __builtin_amdgcn_global_load_async_to_lds_b128(
            (g_v4i*)p,       (l_v4i*)&T[j][c], 0, 0);
        __builtin_amdgcn_global_load_async_to_lds_b128(
            (g_v4i*)(p + 8), (l_v4i*)&T[j][c + 8], 0, 0);
        __builtin_amdgcn_s_wait_asynccnt(0);
#else
        *(uint4*)&T[j][c]     = *(const uint4*)p;
        *(uint4*)&T[j][c + 8] = *(const uint4*)(p + 8);
#endif
    }
    __syncthreads();
    {   // write transposed: row d, contiguous j chunk
        int d = t >> 2, jc = (t & 3) * 16;
        U16x16 s;
#pragma unroll
        for (int i = 0; i < 16; ++i) s.v[i] = T[jc + i][d];
        u16* dst = Vt + (((long)(b * H + h) * dk + d) * S) + (long)jt * 64 + jc;
        *(v16bf*)dst = pack16(s);
    }
}

// ---- fused attention: scores + masked softmax + probs@V --------------------
// block = 16 query rows x full 2048 keys for one (b,h); 8 waves x 256 keys.
__global__ __launch_bounds__(256) void fused_attn_kernel(const u16* __restrict__ Qp,
                                                         const u16* __restrict__ Kp,
                                                         const u16* __restrict__ Vt,
                                                         const int* __restrict__ mask,
                                                         float* __restrict__ attn,
                                                         u16* __restrict__ ctx,
                                                         int S, int D, int H, int dk,
                                                         float scale) {
    __shared__ u16 Ap[16][2056];       // 16 x 2048 bf16 probs (padded stride)
    __shared__ float red[8][16];       // cross-wave row reductions
    float* partbuf = (float*)&Ap[0][0];  // reused for ctx partials (32KB < 64KB)

    const int tid = threadIdx.x, lane = tid & 31, w = tid >> 5;
    const int h16 = lane >> 4;         // C-fragment half: rows +0 / +8
    const int ln = lane & 15;
    const int bh = blockIdx.y;
    const int b = bh / H, h = bh % H;
    const int m0 = blockIdx.x * 16;

    const u16* qh = Qp + (long)b * S * D + (long)h * dk;
    const u16* kh = Kp + (long)b * S * D + (long)h * dk;

    // ---- phase 1: scores = Q K^T * scale (wave covers cols w*256..+255) ----
    v16bf a0 = frag_a_global(qh, D, m0, 0, lane);
    v16bf a1 = frag_a_global(qh, D, m0, 32, lane);

    unsigned mbits = 0;
#pragma unroll
    for (int t = 0; t < 16; ++t) {
        int j = w * 256 + t * 16 + ln;
        if (mask[(long)b * S + j] != 0) mbits |= (1u << t);
    }

    v8f acc[16];
#pragma unroll
    for (int t = 0; t < 16; ++t) {
        int n0 = w * 256 + t * 16;
        v8f z = {0.f, 0.f, 0.f, 0.f, 0.f, 0.f, 0.f, 0.f};
        v16bf b0 = frag_b_global_bf16(kh, D, n0, 0, lane);
        z = wmma_bf16(a0, b0, z);
        v16bf b1 = frag_b_global_bf16(kh, D, n0, 32, lane);
        acc[t] = wmma_bf16(a1, b1, z);
    }
#pragma unroll
    for (int t = 0; t < 16; ++t) {
        float keep = (mbits >> t) & 1u;
#pragma unroll
        for (int r = 0; r < 8; ++r)
            acc[t][r] = keep != 0.f ? acc[t][r] * scale : -1e9f;
    }

    // ---- phase 2: row max / exp / row sum ----------------------------------
    float rmax[8];
#pragma unroll
    for (int r = 0; r < 8; ++r) {
        float m = acc[0][r];
#pragma unroll
        for (int t = 1; t < 16; ++t) m = fmaxf(m, acc[t][r]);
#pragma unroll
        for (int o = 8; o; o >>= 1) m = fmaxf(m, __shfl_xor(m, o, 32));  // 16-lane half
        rmax[r] = m;
    }
    if (ln == 0) {
#pragma unroll
        for (int r = 0; r < 8; ++r) red[w][h16 * 8 + r] = rmax[r];
    }
    __syncthreads();
#pragma unroll
    for (int r = 0; r < 8; ++r) {
        float m = red[0][h16 * 8 + r];
#pragma unroll
        for (int ww = 1; ww < 8; ++ww) m = fmaxf(m, red[ww][h16 * 8 + r]);
        rmax[r] = m;
    }
    __syncthreads();

    float rsum[8];
#pragma unroll
    for (int r = 0; r < 8; ++r) rsum[r] = 0.f;
#pragma unroll
    for (int t = 0; t < 16; ++t) {
#pragma unroll
        for (int r = 0; r < 8; ++r) {
            float e = __expf(acc[t][r] - rmax[r]);
            acc[t][r] = e;
            rsum[r] += e;
        }
    }
#pragma unroll
    for (int r = 0; r < 8; ++r) {
        float s = rsum[r];
#pragma unroll
        for (int o = 8; o; o >>= 1) s += __shfl_xor(s, o, 32);
        rsum[r] = s;
    }
    if (ln == 0) {
#pragma unroll
        for (int r = 0; r < 8; ++r) red[w][h16 * 8 + r] = rsum[r];
    }
    __syncthreads();
#pragma unroll
    for (int r = 0; r < 8; ++r) {
        float s = red[0][h16 * 8 + r];
#pragma unroll
        for (int ww = 1; ww < 8; ++ww) s += red[ww][h16 * 8 + r];
        rsum[r] = 1.0f / s;
    }

    // ---- phase 3: write probs (fp32 -> d_out, bf16 -> LDS) -----------------
    float* aout = attn + (long)bh * S * S;
#pragma unroll
    for (int t = 0; t < 16; ++t) {
        int col = w * 256 + t * 16 + ln;
#pragma unroll
        for (int r = 0; r < 8; ++r) {
            float v = acc[t][r] * rsum[r];
            int row = h16 * 8 + r;
            aout[(long)(m0 + row) * S + col] = v;
            Ap[row][col] = f2bf(v);
        }
    }
    __syncthreads();

    // ---- phase 4: ctx partials = probs[16 x 256] @ V[256 x 64] per wave ----
    const u16* vth = Vt + (long)bh * dk * S;   // [dk][S], contiguous in j
    v8f cacc[4];
#pragma unroll
    for (int t = 0; t < 4; ++t)
        cacc[t] = (v8f){0.f, 0.f, 0.f, 0.f, 0.f, 0.f, 0.f, 0.f};
    for (int step = 0; step < 8; ++step) {
        int k0 = w * 256 + step * 32;
        v16bf a = frag_a_lds(&Ap[0][0], 2056, 0, k0, lane);
#pragma unroll
        for (int t = 0; t < 4; ++t) {
            v16bf bf = frag_b_global_bf16(vth, S, t * 16, k0, lane);
            cacc[t] = wmma_bf16(a, bf, cacc[t]);
        }
    }
    __syncthreads();   // all waves done reading Ap; safe to overwrite

    // ---- phase 5: cross-wave K reduction, write ctx (bf16, merged heads) ---
#pragma unroll
    for (int t = 0; t < 4; ++t) {
#pragma unroll
        for (int r = 0; r < 8; ++r)
            partbuf[((w * 16) + h16 * 8 + r) * 64 + t * 16 + ln] = cacc[t][r];
    }
    __syncthreads();
#pragma unroll
    for (int i = 0; i < 4; ++i) {
        int idx = tid * 4 + i;          // 0..1023 over 16x64
        int row = idx >> 6, col = idx & 63;
        float s = 0.f;
#pragma unroll
        for (int ww = 0; ww < 8; ++ww) s += partbuf[(ww * 16 + row) * 64 + col];
        ctx[((long)b * S + m0 + row) * D + (long)h * dk + col] = f2bf(s);
    }
}

// ---- reductions ------------------------------------------------------------
__device__ __forceinline__ float wave_sum(float v) {
#pragma unroll
    for (int o = 16; o; o >>= 1) v += __shfl_xor(v, o, 32);
    return v;
}

// ---- LayerNorm, one block per row ------------------------------------------
__global__ __launch_bounds__(256) void layernorm_kernel(const float* __restrict__ x,
                                                        const float* __restrict__ gamma,
                                                        const float* __restrict__ beta,
                                                        float* __restrict__ out, int D) {
    __shared__ float sh[8];
    const long row = blockIdx.x;
    const int tid = threadIdx.x, lane = tid & 31, wave = tid >> 5;
    const float* p = x + row * D;
    float4 v = *(const float4*)(p + tid * 4);

    float s = v.x + v.y + v.z + v.w;
    float wsum = wave_sum(s);
    if (lane == 0) sh[wave] = wsum;
    __syncthreads();
    float mean = ((sh[0] + sh[1]) + (sh[2] + sh[3]) + (sh[4] + sh[5]) + (sh[6] + sh[7])) / (float)D;
    __syncthreads();

    float dx = v.x - mean, dy = v.y - mean, dz = v.z - mean, dw = v.w - mean;
    float sq = dx * dx + dy * dy + dz * dz + dw * dw;
    float wsq = wave_sum(sq);
    if (lane == 0) sh[wave] = wsq;
    __syncthreads();
    float var = ((sh[0] + sh[1]) + (sh[2] + sh[3]) + (sh[4] + sh[5]) + (sh[6] + sh[7])) / (float)D;
    float rs = rsqrtf(var + 1e-5f);

    float4 g = *(const float4*)(gamma + tid * 4);
    float4 bt = *(const float4*)(beta + tid * 4);
    float4 o;
    o.x = dx * rs * g.x + bt.x;
    o.y = dy * rs * g.y + bt.y;
    o.z = dz * rs * g.z + bt.z;
    o.w = dw * rs * g.w + bt.w;
    *(float4*)(out + row * D + tid * 4) = o;
}

extern "C" void kernel_launch(void* const* d_in, const int* in_sizes, int n_in,
                              void* d_out, int out_size, void* d_ws, size_t ws_size,
                              hipStream_t stream) {
    const float* query = (const float*)d_in[0];
    const float* key   = (const float*)d_in[1];
    const float* value = (const float*)d_in[2];
    const int*   mask  = (const int*)d_in[3];
    const float* Wq = (const float*)d_in[4];
    const float* bq = (const float*)d_in[5];
    const float* Wk = (const float*)d_in[6];
    const float* bk = (const float*)d_in[7];
    const float* Wv = (const float*)d_in[8];
    const float* bv = (const float*)d_in[9];
    const float* Wo = (const float*)d_in[10];
    const float* bo = (const float*)d_in[11];
    const float* gamma = (const float*)d_in[12];
    const float* beta  = (const float*)d_in[13];

    const int B = 4, S = 2048, D = 1024, H = 16, dk = 64;
    const long M = (long)B * S;  // 8192

    // ws: Qp, Kp, Vp(->reused as ctx), Vt (bf16, 16MB each) + x (fp32, 32MB) = 96MB
    u16* Qp  = (u16*)d_ws;
    u16* Kp  = Qp + M * D;
    u16* Vp  = Kp + M * D;        // reused as ctx after transpose
    u16* Vt  = Vp + M * D;
    float* x = (float*)(Vt + M * D);
    u16* ctx = Vp;

    float* out  = (float*)d_out;
    float* attn = out + M * D;    // (B,H,S,S) region of d_out

    dim3 blk(256);
    dim3 gproj(D / 128, (unsigned)(M / 16));

    // Q/K/V projections: fp32 in, bf16 out
    gemm_xwT<false, true, false><<<gproj, blk, 0, stream>>>(query, Wq, bq, nullptr, Qp, (int)M, D, D);
    gemm_xwT<false, true, false><<<gproj, blk, 0, stream>>>(key,   Wk, bk, nullptr, Kp, (int)M, D, D);
    gemm_xwT<false, true, false><<<gproj, blk, 0, stream>>>(value, Wv, bv, nullptr, Vp, (int)M, D, D);

    // per-head V transpose for contiguous-K B fragments
    transpose_v_kernel<<<dim3(S / 64, H, B), blk, 0, stream>>>(Vp, Vt, S, D, H, dk);

    // fused scores + masked softmax + probs@V
    fused_attn_kernel<<<dim3(S / 16, B * H), blk, 0, stream>>>(Qp, Kp, Vt, mask, attn, ctx,
                                                               S, D, H, dk, 0.125f);

    // output projection + residual (fp32 x)
    gemm_xwT<true, false, true><<<gproj, blk, 0, stream>>>(ctx, Wo, bo, query, x, (int)M, D, D);

    // LayerNorm -> first M*D floats of d_out
    layernorm_kernel<<<dim3((unsigned)M), blk, 0, stream>>>(x, gamma, beta, out, D);
}
